// EncoderBlock_11819749998915
// MI455X (gfx1250) — matrix operand order repaired and checked by reading, not compiled, not run on hardware
//
#include <hip/hip_runtime.h>

// ---------------------------------------------------------------------------
// Transformer encoder block for MI455X (gfx1250), wave32, WMMA f16 path.
// B=2 S=2048 D=1024 H=16 DK=64 FF=4096.  M = B*S = 4096 rows everywhere.
// GEMMs: double-buffered LDS fed by async global->LDS (ASYNCcnt) copies.
// ---------------------------------------------------------------------------

#define Bc 2
#define Sc 2048
#define Dc 1024
#define Hc 16
#define DKc 64
#define FFc 4096
#define Mc (Bc * Sc)     // 4096

typedef __attribute__((ext_vector_type(16))) _Float16 v16h;
typedef __attribute__((ext_vector_type(8)))  _Float16 v8h;
typedef __attribute__((ext_vector_type(4)))  _Float16 v4h;
typedef __attribute__((ext_vector_type(8)))  float    v8f;
typedef __attribute__((ext_vector_type(4)))  float    v4f;

static __device__ __forceinline__ v16h concat8(v8h lo, v8h hi) {
  return __builtin_shufflevector(lo, hi, 0,1,2,3,4,5,6,7,8,9,10,11,12,13,14,15);
}

// A-fragment (16x32 f16, MxK): lanes 0-15 hold M=lr,K={0..7,16..23}+ks ;
// lanes 16-31 hold M=lr,K={8..15,24..31}+ks.  colA = ks + hb*8.
static __device__ __forceinline__ v16h load_a_frag(const _Float16* row, int colA) {
  v8h lo = *(const v8h*)(row + colA);
  v8h hi = *(const v8h*)(row + colA + 16);
  return concat8(lo, hi);
}

// B-fragment (32x16 f16, KxN): lanes 0-15 hold N=lr,K={0..7,16..23}+ks ;
// lanes 16-31 hold N=lr,K={8..15,24..31}+ks.  p = row + ks + hb*8.
static __device__ __forceinline__ v16h load_b_frag(const _Float16* p) {
  v8h lo = *(const v8h*)(p);
  v8h hi = *(const v8h*)(p + 16);
  return concat8(lo, hi);
}

// Async global->LDS copy of 16 bytes per lane (CDNA5, tracked by ASYNCcnt).
// lds_off = wave-relative LDS byte offset (generic LDS pointer truncated to 32b,
// per ISA 10.2).
static __device__ __forceinline__ void async_copy16(unsigned lds_off, const void* gptr) {
  asm volatile("global_load_async_to_lds_b128 %0, %1, off"
               :: "v"(lds_off), "v"((unsigned long long)(uintptr_t)gptr)
               : "memory");
}

// ---------------------------------------------------------------------------
// f32 -> f16 cast (weights), 4 elements/thread
// ---------------------------------------------------------------------------
__global__ void cast_kernel(const float* __restrict__ in, _Float16* __restrict__ out, int n4) {
  int i = blockIdx.x * blockDim.x + threadIdx.x;
  if (i < n4) {
    v4f f = ((const v4f*)in)[i];
    ((v4h*)out)[i] = __builtin_convertvector(f, v4h);
  }
}

// ---------------------------------------------------------------------------
// LayerNorm (faithful: unbiased std, sqrt(std + eps)), f32 in -> f16 out.
// One 256-thread block per row of 1024.
// ---------------------------------------------------------------------------
__global__ __launch_bounds__(256) void ln_kernel(const float* __restrict__ x,
                                                 const float* __restrict__ alpha,
                                                 const float* __restrict__ beta,
                                                 _Float16* __restrict__ out) {
  const int row = blockIdx.x;
  const int tid = threadIdx.x;
  const float* xr = x + (size_t)row * Dc;
  float v[4];
#pragma unroll
  for (int i = 0; i < 4; i++) v[i] = xr[tid + i * 256];

  __shared__ float red[8];
  float s = v[0] + v[1] + v[2] + v[3];
#pragma unroll
  for (int off = 16; off >= 1; off >>= 1) s += __shfl_xor(s, off);
  if ((tid & 31) == 0) red[tid >> 5] = s;
  __syncthreads();
  float tot = 0.f;
#pragma unroll
  for (int i = 0; i < 8; i++) tot += red[i];
  const float mean = tot * (1.0f / (float)Dc);

  float q = 0.f;
#pragma unroll
  for (int i = 0; i < 4; i++) { float d = v[i] - mean; q += d * d; }
  __syncthreads();
#pragma unroll
  for (int off = 16; off >= 1; off >>= 1) q += __shfl_xor(q, off);
  if ((tid & 31) == 0) red[tid >> 5] = q;
  __syncthreads();
  float qt = 0.f;
#pragma unroll
  for (int i = 0; i < 8; i++) qt += red[i];

  const float var  = qt * (1.0f / (float)(Dc - 1));
  const float stdv = sqrtf(var);
  const float scl  = alpha[0] * rsqrtf(stdv + 1e-6f);
  const float b0   = beta[0];
  _Float16* orow = out + (size_t)row * Dc;
#pragma unroll
  for (int i = 0; i < 4; i++)
    orow[tid + i * 256] = (_Float16)((v[i] - mean) * scl + b0);
}

// ---------------------------------------------------------------------------
// GEMM: C[M,N] = act(A[M,K] * W[N,K]^T + bias[N]) (+ res[M,N]),  f16 WMMA.
// Block 256 (8 waves), tile 128x128, K-step 64. Wave = 64x32 (4x2 subtiles).
// LDS double-buffered; tiles prefetched with async global->LDS copies.
// ---------------------------------------------------------------------------
template <bool RELU, bool RES, bool OUTF16>
__global__ __launch_bounds__(256) void gemm_kernel(const _Float16* __restrict__ A,
                                                   const _Float16* __restrict__ W,
                                                   const float* __restrict__ bias,
                                                   const float* __restrict__ res,
                                                   void* __restrict__ out,
                                                   int M, int N, int Kd) {
  __shared__ alignas(16) _Float16 As[2][128][72];
  __shared__ alignas(16) _Float16 Bs[2][128][72];

  const int tid = threadIdx.x;
  const int wid = tid >> 5;
  const int lane = tid & 31;
  const int lr = lane & 15;
  const int hb = lane >> 4;
  const int wave_m = wid >> 2;   // 0..1
  const int wave_n = wid & 3;    // 0..3
  const int m0 = blockIdx.x * 128;
  const int n0 = blockIdx.y * 128;

  const int ldr = tid >> 1;            // 0..127 (tile row loaded by this thread)
  const int ldc = (tid & 1) * 32;      // 0 or 32

  const _Float16* gA = A + (size_t)(m0 + ldr) * Kd + ldc;
  const _Float16* gB = W + (size_t)(n0 + ldr) * Kd + ldc;

  v8f acc[4][2] = {};

  // issue tile 0 into buffer 0 (4 x 16B for A, 4 x 16B for B per thread)
  {
    const unsigned la = (unsigned)(uintptr_t)&As[0][ldr][ldc];
    const unsigned lb = (unsigned)(uintptr_t)&Bs[0][ldr][ldc];
#pragma unroll
    for (int q = 0; q < 4; q++) {
      async_copy16(la + q * 16, gA + q * 8);
      async_copy16(lb + q * 16, gB + q * 8);
    }
  }

  const int T = Kd >> 6;  // K-tiles of 64
  for (int t = 0; t < T; ++t) {
    const int cur = t & 1;
    if (t + 1 < T) {
      // prefetch next tile into the other buffer
      const int k1 = (t + 1) << 6;
      const unsigned la = (unsigned)(uintptr_t)&As[cur ^ 1][ldr][ldc];
      const unsigned lb = (unsigned)(uintptr_t)&Bs[cur ^ 1][ldr][ldc];
#pragma unroll
      for (int q = 0; q < 4; q++) {
        async_copy16(la + q * 16, gA + k1 + q * 8);
        async_copy16(lb + q * 16, gB + k1 + q * 8);
      }
      // current tile's 8 copies done; the 8 just-issued may remain in flight
      asm volatile("s_wait_asynccnt 0x8" ::: "memory");
    } else {
      asm volatile("s_wait_asynccnt 0x0" ::: "memory");
    }
    __syncthreads();

#pragma unroll
    for (int ks = 0; ks < 64; ks += 32) {
      v16h a[4], b[2];
#pragma unroll
      for (int i = 0; i < 4; i++)
        a[i] = load_a_frag(&As[cur][wave_m * 64 + i * 16 + lr][0], ks + hb * 8);
#pragma unroll
      for (int j = 0; j < 2; j++)
        b[j] = load_b_frag(&Bs[cur][wave_n * 32 + j * 16 + lr][ks + hb * 8]);
#pragma unroll
      for (int i = 0; i < 4; i++)
#pragma unroll
        for (int j = 0; j < 2; j++)
          acc[i][j] = __builtin_amdgcn_wmma_f32_16x16x32_f16(
              false, a[i], false, b[j], (short)0, acc[i][j], false, false);
    }
    __syncthreads();  // all waves done reading `cur` before it is refilled
  }

  // epilogue: C layout = VGPR r, lanes<16 -> M=r, lanes>=16 -> M=8+r, N=lr
  const int mb = m0 + wave_m * 64;
  const int nb = n0 + wave_n * 32;
#pragma unroll
  for (int j = 0; j < 2; j++) {
    const int col = nb + j * 16 + lr;
    const float bj = bias[col];
#pragma unroll
    for (int i = 0; i < 4; i++) {
      const int rowb = mb + i * 16 + hb * 8;
#pragma unroll
      for (int r = 0; r < 8; r++) {
        const size_t idx = (size_t)(rowb + r) * N + col;
        float v = acc[i][j][r] + bj;
        if (RELU) v = fmaxf(v, 0.0f);
        if (RES)  v += res[idx];
        if (OUTF16) ((_Float16*)out)[idx] = (_Float16)v;
        else        ((float*)out)[idx] = v;
      }
    }
  }
}

// ---------------------------------------------------------------------------
// Flash attention: grid (S/64, B*H), block 128 (4 waves, 16 q-rows each).
// Q,K,V,O all (B,S,D) f16, head h at column offset h*64. KV blocks of 64.
// ---------------------------------------------------------------------------
__global__ __launch_bounds__(128) void attn_kernel(const _Float16* __restrict__ Q,
                                                   const _Float16* __restrict__ K,
                                                   const _Float16* __restrict__ V,
                                                   _Float16* __restrict__ O) {
  __shared__ alignas(16) _Float16 Kb[64][72];      // [kv][dk]
  __shared__ alignas(16) _Float16 Vt[64][72];      // [dk][kv] (transposed)
  __shared__ alignas(16) _Float16 Pst[4][16][72];  // per-wave P staging

  const int tid = threadIdx.x;
  const int wid = tid >> 5;
  const int lane = tid & 31;
  const int lr = lane & 15;
  const int hb = lane >> 4;

  const int b = blockIdx.y / Hc;
  const int h = blockIdx.y % Hc;
  const size_t base = (size_t)b * Sc * Dc + (size_t)h * DKc;
  const int q0 = blockIdx.x * 64;

  // Q fragments for this wave's 16 rows (K=64 -> two 32-wide k-steps)
  const _Float16* qrow = Q + base + (size_t)(q0 + wid * 16 + lr) * Dc;
  const v16h qf0 = load_a_frag(qrow, 0 + hb * 8);
  const v16h qf1 = load_a_frag(qrow, 32 + hb * 8);

  float mi[8], li[8];
#pragma unroll
  for (int r = 0; r < 8; r++) { mi[r] = -1e30f; li[r] = 0.f; }
  v8f oacc[4] = {};

  const int srow = tid >> 1;        // kv row staged by this thread
  const int scol = (tid & 1) * 32;  // dk column half

  for (int kv0 = 0; kv0 < Sc; kv0 += 64) {
    // stage K (row-major) and V (transposed)
    {
      const v8h* kp = (const v8h*)(K + base + (size_t)(kv0 + srow) * Dc + scol);
      v8h* ksp = (v8h*)&Kb[srow][scol];
      ksp[0] = kp[0]; ksp[1] = kp[1]; ksp[2] = kp[2]; ksp[3] = kp[3];
      const v8h* vp = (const v8h*)(V + base + (size_t)(kv0 + srow) * Dc + scol);
      v8h vv[4];
#pragma unroll
      for (int q = 0; q < 4; q++) vv[q] = vp[q];
#pragma unroll
      for (int q = 0; q < 4; q++)
#pragma unroll
        for (int e = 0; e < 8; e++)
          Vt[scol + q * 8 + e][srow] = vv[q][e];
    }
    __syncthreads();

    // scores S = Q * K^T  (16 x 64)
    v8f sc[4] = {};
#pragma unroll
    for (int ks = 0; ks < 2; ks++) {
      const v16h aq = ks ? qf1 : qf0;
#pragma unroll
      for (int j = 0; j < 4; j++) {
        v16h bf = load_b_frag(&Kb[j * 16 + lr][ks * 32 + hb * 8]);
        sc[j] = __builtin_amdgcn_wmma_f32_16x16x32_f16(
            false, aq, false, bf, (short)0, sc[j], false, false);
      }
    }
#pragma unroll
    for (int j = 0; j < 4; j++) sc[j] *= 0.125f;  // 1/sqrt(64)

    // online softmax (row's 16 cols live in one 16-lane half)
#pragma unroll
    for (int r = 0; r < 8; r++) {
      float mb = fmaxf(fmaxf(sc[0][r], sc[1][r]), fmaxf(sc[2][r], sc[3][r]));
#pragma unroll
      for (int off = 8; off >= 1; off >>= 1) mb = fmaxf(mb, __shfl_xor(mb, off));
      const float mn = fmaxf(mi[r], mb);
      const float al = __expf(mi[r] - mn);
      float ssum = 0.f;
#pragma unroll
      for (int j = 0; j < 4; j++) {
        float p = __expf(sc[j][r] - mn);
        Pst[wid][r + hb * 8][j * 16 + lr] = (_Float16)p;
        ssum += p;
      }
#pragma unroll
      for (int off = 8; off >= 1; off >>= 1) ssum += __shfl_xor(ssum, off);
      li[r] = li[r] * al + ssum;
      mi[r] = mn;
#pragma unroll
      for (int j = 0; j < 4; j++) oacc[j][r] = oacc[j][r] * al;
    }

    // fence same-wave LDS store -> load (CDNA5 split DS counter)
    asm volatile("s_wait_dscnt 0" ::: "memory");

    // O += P * V  (K dim = kv block of 64 -> two k-steps)
    const _Float16* prow = &Pst[wid][lr][0];
#pragma unroll
    for (int ks = 0; ks < 2; ks++) {
      v16h ap = load_a_frag(prow, ks * 32 + hb * 8);
#pragma unroll
      for (int j = 0; j < 4; j++) {
        v16h bv = load_b_frag(&Vt[j * 16 + lr][ks * 32 + hb * 8]);
        oacc[j] = __builtin_amdgcn_wmma_f32_16x16x32_f16(
            false, ap, false, bv, (short)0, oacc[j], false, false);
      }
    }
    __syncthreads();
  }

  // normalize + write
#pragma unroll
  for (int r = 0; r < 8; r++) {
    const float inv = 1.0f / li[r];
    const int row = q0 + wid * 16 + hb * 8 + r;
    _Float16* orow = O + base + (size_t)row * Dc;
#pragma unroll
    for (int j = 0; j < 4; j++)
      orow[j * 16 + lr] = (_Float16)(oacc[j][r] * inv);
  }
}

// ---------------------------------------------------------------------------
// Host-side orchestration
// ---------------------------------------------------------------------------
extern "C" void kernel_launch(void* const* d_in, const int* in_sizes, int n_in,
                              void* d_out, int out_size, void* d_ws, size_t ws_size,
                              hipStream_t stream) {
  (void)in_sizes; (void)n_in; (void)out_size; (void)ws_size;

  const float* x    = (const float*)d_in[0];
  const float* wq   = (const float*)d_in[1];
  const float* bq   = (const float*)d_in[2];
  const float* wk   = (const float*)d_in[3];
  const float* bk   = (const float*)d_in[4];
  const float* wv   = (const float*)d_in[5];
  const float* bv   = (const float*)d_in[6];
  const float* wo   = (const float*)d_in[7];
  const float* bo   = (const float*)d_in[8];
  const float* w1   = (const float*)d_in[9];
  const float* b1   = (const float*)d_in[10];
  const float* w2   = (const float*)d_in[11];
  const float* b2   = (const float*)d_in[12];
  const float* ln1a = (const float*)d_in[13];
  const float* ln1b = (const float*)d_in[14];
  const float* ln2a = (const float*)d_in[15];
  const float* ln2b = (const float*)d_in[16];

  char* ws = (char*)d_ws;
  const size_t MB = 1024 * 1024;
  _Float16* wq_h = (_Float16*)(ws + 0);        // 2MB
  _Float16* wk_h = (_Float16*)(ws + 2 * MB);   // 2MB
  _Float16* wv_h = (_Float16*)(ws + 4 * MB);   // 2MB
  _Float16* wo_h = (_Float16*)(ws + 6 * MB);   // 2MB
  _Float16* w1_h = (_Float16*)(ws + 8 * MB);   // 8MB
  _Float16* w2_h = (_Float16*)(ws + 16 * MB);  // 8MB
  _Float16* xn   = (_Float16*)(ws + 24 * MB);  // 8MB
  _Float16* d_q  = (_Float16*)(ws + 32 * MB);  // 8MB
  _Float16* d_k  = (_Float16*)(ws + 40 * MB);  // 8MB
  _Float16* d_v  = (_Float16*)(ws + 48 * MB);  // 8MB
  _Float16* attn = (_Float16*)(ws + 56 * MB);  // 8MB
  float*    hbuf = (float*)   (ws + 64 * MB);  // 16MB
  _Float16* hn   = (_Float16*)(ws + 80 * MB);  // 8MB
  _Float16* f1   = (_Float16*)(ws + 88 * MB);  // 32MB

  const dim3 blk256(256);

  // cast weights to f16
  {
    const int nDD4 = (Dc * Dc) / 4;      // 262144
    const int nDF4 = (Dc * FFc) / 4;     // 1048576
    cast_kernel<<<dim3((nDD4 + 255) / 256), blk256, 0, stream>>>(wq, wq_h, nDD4);
    cast_kernel<<<dim3((nDD4 + 255) / 256), blk256, 0, stream>>>(wk, wk_h, nDD4);
    cast_kernel<<<dim3((nDD4 + 255) / 256), blk256, 0, stream>>>(wv, wv_h, nDD4);
    cast_kernel<<<dim3((nDD4 + 255) / 256), blk256, 0, stream>>>(wo, wo_h, nDD4);
    cast_kernel<<<dim3((nDF4 + 255) / 256), blk256, 0, stream>>>(w1, w1_h, nDF4);
    cast_kernel<<<dim3((nDF4 + 255) / 256), blk256, 0, stream>>>(w2, w2_h, nDF4);
  }

  // ln1(x) -> xn (f16)
  ln_kernel<<<dim3(Mc), blk256, 0, stream>>>(x, ln1a, ln1b, xn);

  // Q, K, V projections (f16 out)
  gemm_kernel<false, false, true><<<dim3(Mc / 128, Dc / 128), blk256, 0, stream>>>(
      xn, wq_h, bq, nullptr, d_q, Mc, Dc, Dc);
  gemm_kernel<false, false, true><<<dim3(Mc / 128, Dc / 128), blk256, 0, stream>>>(
      xn, wk_h, bk, nullptr, d_k, Mc, Dc, Dc);
  gemm_kernel<false, false, true><<<dim3(Mc / 128, Dc / 128), blk256, 0, stream>>>(
      xn, wv_h, bv, nullptr, d_v, Mc, Dc, Dc);

  // flash attention -> attn (f16)
  attn_kernel<<<dim3(Sc / 64, Bc * Hc), dim3(128), 0, stream>>>(d_q, d_k, d_v, attn);

  // h = x + attn * wo^T + bo  (f32)
  gemm_kernel<false, true, false><<<dim3(Mc / 128, Dc / 128), blk256, 0, stream>>>(
      attn, wo_h, bo, x, hbuf, Mc, Dc, Dc);

  // ln2(h) -> hn (f16)
  ln_kernel<<<dim3(Mc), blk256, 0, stream>>>(hbuf, ln2a, ln2b, hn);

  // f1 = relu(hn * w1^T + b1) (f16)
  gemm_kernel<true, false, true><<<dim3(Mc / 128, FFc / 128), blk256, 0, stream>>>(
      hn, w1_h, b1, nullptr, f1, Mc, FFc, Dc);

  // out = h + f1 * w2^T + b2  (f32)
  gemm_kernel<false, true, false><<<dim3(Mc / 128, Dc / 128), blk256, 0, stream>>>(
      f1, w2_h, b2, hbuf, d_out, Mc, Dc, FFc);
}
